// RelativeGlobalAttention_49091476194107
// MI455X (gfx1250) — matrix operand order repaired and checked by reading, not compiled, not run on hardware
//
#include <hip/hip_runtime.h>
#include <hip/hip_bf16.h>

// ---------------------------------------------------------------------------
// Relative Global Attention for MI455X (gfx1250, wave32, WMMA bf16->f32)
//   B=2, L=2048, D=1024, H=16, DEPTH=64, MAX_REL=2048
// d_out = [ out (B*L*D f32) | weights (B*H*L*L f32) ]
// ---------------------------------------------------------------------------

#define BB 2
#define HH 16
#define LL 2048
#define DD 1024
#define DEPTHD 64

typedef __attribute__((ext_vector_type(16))) __bf16 v16bf;
typedef __attribute__((ext_vector_type(8)))  __bf16 bf16x8;
typedef __attribute__((ext_vector_type(8)))  float  v8f;
typedef __attribute__((ext_vector_type(4)))  float  f32x4;

// D = A(16x32 bf16) * B(32x16 bf16) + C(16x16 f32)
__device__ __forceinline__ v8f wmma_bf16(v16bf a, v16bf b, v8f c) {
  return __builtin_amdgcn_wmma_f32_16x16x32_bf16(false, a, false, b, (short)0, c,
                                                 false, false);
}

__device__ __forceinline__ float bperm_f(int srclane, float x) {
  return __builtin_bit_cast(
      float, __builtin_amdgcn_ds_bpermute(srclane << 2, __builtin_bit_cast(int, x)));
}

// B-fragment: lane holds col = lane&15, K chunks [p..p+7] and [p+16..p+23]
// where p already includes (lane>>4)*8 (16B-aligned).
__device__ __forceinline__ v16bf ld_bfrag16(const __bf16* p) {
  bf16x8 c0 = *(const bf16x8*)p;
  bf16x8 c1 = *(const bf16x8*)(p + 16);
  v16bf r;
#pragma unroll
  for (int i = 0; i < 8; ++i) { r[i] = c0[i]; r[8 + i] = c1[i]; }
  return r;
}

// A-fragment: lane row = lane&15, K chunks [kb..kb+7] and [kb+16..kb+23]
// where kb already includes (lane>>4)*8.
__device__ __forceinline__ v16bf ld_afrag_bf(const __bf16* rowp, int kb) {
  bf16x8 c0 = *(const bf16x8*)(rowp + kb);
  bf16x8 c1 = *(const bf16x8*)(rowp + kb + 16);
  v16bf r;
#pragma unroll
  for (int i = 0; i < 8; ++i) { r[i] = c0[i]; r[8 + i] = c1[i]; }
  return r;
}

// f32 -> bf16 A-fragment (chunks at kb and kb+16, kb includes half*8)
__device__ __forceinline__ v16bf cvt_afrag_f32(const float* rowp, int kb) {
  f32x4 a0 = *(const f32x4*)(rowp + kb);
  f32x4 a1 = *(const f32x4*)(rowp + kb + 4);
  f32x4 a2 = *(const f32x4*)(rowp + kb + 16);
  f32x4 a3 = *(const f32x4*)(rowp + kb + 20);
  v16bf r;
#pragma unroll
  for (int i = 0; i < 4; ++i) {
    r[i]      = (__bf16)a0[i];
    r[4 + i]  = (__bf16)a1[i];
    r[8 + i]  = (__bf16)a2[i];
    r[12 + i] = (__bf16)a3[i];
  }
  return r;
}

// f32 -> bf16 B-fragment: chunks at p and p+16 (p includes half*8)
__device__ __forceinline__ v16bf cvt_bfrag_f32(const float* p) {
  f32x4 a0 = *(const f32x4*)(p);
  f32x4 a1 = *(const f32x4*)(p + 4);
  f32x4 a2 = *(const f32x4*)(p + 16);
  f32x4 a3 = *(const f32x4*)(p + 20);
  v16bf r;
#pragma unroll
  for (int i = 0; i < 4; ++i) {
    r[i]      = (__bf16)a0[i];
    r[4 + i]  = (__bf16)a1[i];
    r[8 + i]  = (__bf16)a2[i];
    r[12 + i] = (__bf16)a3[i];
  }
  return r;
}

// ---------------------------------------------------------------------------
// Kernel 1: rel_emb f32 -> bf16
// ---------------------------------------------------------------------------
__global__ void k_cvt_rel(const float* __restrict__ rel, __bf16* __restrict__ relb,
                          int n) {
  int i = blockIdx.x * blockDim.x + threadIdx.x;
  if (i < n) relb[i] = (__bf16)rel[i];
}

// ---------------------------------------------------------------------------
// Kernel 2: QKV projection  X @ Wq^T + bq  -> split-head bf16 [B,H,L,64]
// grid: (128, 8, 3)  block: 256 (8 waves; each wave: one 16-row A strip x
// two 16x16 N tiles => two independent WMMA accumulator chains to cover the
// bf16 WMMA->WMMA RAW hazard slots)
// ---------------------------------------------------------------------------
__global__ __launch_bounds__(256) void k_proj_qkv(
    const float* __restrict__ q_in, const float* __restrict__ k_in,
    const float* __restrict__ v_in, const float* __restrict__ Wq,
    const float* __restrict__ bq, __bf16* __restrict__ qkv_out) {
  const int tid = threadIdx.x;
  const int lane = tid & 31, wave = tid >> 5;
  const int half = lane >> 4, ln = lane & 15;
  const int mt = blockIdx.x * 2 + (wave >> 2);
  const int n0 = (blockIdx.y * 8 + (wave & 3) * 2) * 16;
  const int n1 = n0 + 16;
  const int z = blockIdx.z;
  const float* X = (z == 0) ? q_in : (z == 1) ? k_in : v_in;
  __bf16* outz = qkv_out + (size_t)z * ((size_t)BB * LL * DD);

  const int m0 = mt * 16;
  const float* xrow = X + (size_t)(m0 + ln) * DD;
  const float* w0 = Wq + (size_t)(n0 + ln) * DD;
  const float* w1 = Wq + (size_t)(n1 + ln) * DD;

  v8f c0 = {}, c1 = {};
#pragma unroll 2
  for (int ks = 0; ks < DD / 32; ++ks) {
    const int kb = ks * 32;
    v16bf a  = cvt_afrag_f32(xrow, kb + half * 8);
    v16bf b0 = cvt_bfrag_f32(w0 + kb + half * 8);
    v16bf b1 = cvt_bfrag_f32(w1 + kb + half * 8);
    c0 = wmma_bf16(a, b0, c0);   // independent chains -> no hazard stall
    c1 = wmma_bf16(a, b1, c1);
  }

#pragma unroll
  for (int t = 0; t < 2; ++t) {
    const v8f& c = t ? c1 : c0;
    const int n = (t ? n1 : n0) + ln;
    const float bias = bq[n];
    const int h = n >> 6, d = n & 63;
#pragma unroll
    for (int r = 0; r < 8; ++r) {
      const int row = r + half * 8;
      const int m = m0 + row;
      const int bb = m >> 11, l = m & (LL - 1);
      outz[(((size_t)bb * HH + h) * LL + l) * DEPTHD + d] = (__bf16)(c[r] + bias);
    }
  }
}

// ---------------------------------------------------------------------------
// Kernel 3: attention core. One block = 16 q-rows of one (b,h).
// LDS: logits f32[16][2048] | wbf bf16[16][2048] | vt bf16[64][72] | reductions
// ---------------------------------------------------------------------------
#define VT_STRIDE 72                 // 144B rows: keeps every row 16B aligned
#define SM_LOGITS 0
#define SM_WBF    131072
#define SM_VT     196608
#define SM_RED    205824
#define SM_RMAX   206848
#define SM_RINV   206912
#define SM_CRED   206976
#define SM_TOTAL  211072

__global__ __launch_bounds__(256) void k_attn(
    const __bf16* __restrict__ qh, const __bf16* __restrict__ kh,
    const __bf16* __restrict__ vh, const __bf16* __restrict__ relb,
    float* __restrict__ wout, __bf16* __restrict__ oh) {
  extern __shared__ char smem[];
  float*  logits = (float*)(smem + SM_LOGITS);   // [16][2048]
  __bf16* wbf    = (__bf16*)(smem + SM_WBF);     // [16][2048]
  __bf16* vt     = (__bf16*)(smem + SM_VT);      // [64][VT_STRIDE]
  float*  red    = (float*)(smem + SM_RED);      // [16][16]
  float*  rmax   = (float*)(smem + SM_RMAX);     // [16]
  float*  rinv   = (float*)(smem + SM_RINV);     // [16]
  float*  cred   = (float*)(smem + SM_CRED);     // [4][256]

  const int tid = threadIdx.x, lane = tid & 31, wave = tid >> 5;
  const int half = lane >> 4, ln = lane & 15;
  const int qt = blockIdx.x;
  const int bh = blockIdx.y;
  const int q0 = qt * 16;
  const int nkt = qt + 1;                       // causal: k-tiles 0..qt
  const int kpad = (nkt * 16 + 63) & ~63;       // round up to V-chunk

  const size_t rowbase = (size_t)bh * LL;

  // Q A-fragments (depth = 64 -> two K=32 steps), shared by all phases.
  const __bf16* qrow = qh + (rowbase + q0 + ln) * DEPTHD;
  const v16bf aq0 = ld_afrag_bf(qrow, 0 + half * 8);
  const v16bf aq1 = ld_afrag_bf(qrow, 32 + half * 8);

  // ---- Phase 1: logits = (Q K^T + skewed Q rel^T) / 8, masked, into LDS ----
  for (int kt = wave; kt < nkt; kt += 8) {
    const int k0 = kt * 16;
    const int kcol = k0 + ln;
    if (kt + 8 < nkt)
      __builtin_prefetch(kh + (rowbase + k0 + 128 + ln) * DEPTHD, 0, 1);

    const __bf16* krow = kh + (rowbase + kcol) * DEPTHD;

    // rel band: value(q,k) = q . rel[2047 + k - q]; per tile need m0-15..m0+15
    const int m0 = (LL - 1) + k0 - q0;
    int mA = m0 - 15 + ln; mA = mA < 0 ? 0 : (mA > LL - 1 ? LL - 1 : mA);
    int mB = m0 + 1 + ln;  mB = mB < 0 ? 0 : (mB > LL - 1 ? LL - 1 : mB);
    const __bf16* ra = relb + (size_t)mA * DEPTHD;
    const __bf16* rb = relb + (size_t)mB * DEPTHD;

    // three independent accumulator chains -> WMMA hazard slots stay filled
    v8f qk = {}, rt0 = {}, rt1 = {};
    qk  = wmma_bf16(aq0, ld_bfrag16(krow + 0 + half * 8), qk);
    rt0 = wmma_bf16(aq0, ld_bfrag16(ra + 0 + half * 8), rt0);
    rt1 = wmma_bf16(aq0, ld_bfrag16(rb + 0 + half * 8), rt1);
    qk  = wmma_bf16(aq1, ld_bfrag16(krow + 32 + half * 8), qk);
    rt0 = wmma_bf16(aq1, ld_bfrag16(ra + 32 + half * 8), rt0);
    rt1 = wmma_bf16(aq1, ld_bfrag16(rb + 32 + half * 8), rt1);

#pragma unroll
    for (int r = 0; r < 8; ++r) {
      const int row = r + half * 8;
      const int c = ln - row + 15;              // 0..30 (skew realignment)
      const int srcl = (lane & 16) | (c & 15);  // same-row element, other lane
      const float v0 = bperm_f(srcl, rt0[r]);
      const float v1 = bperm_f(srcl, rt1[r]);
      const float rv = (c < 16) ? v0 : v1;
      float val = (qk[r] + rv) * 0.125f;        // 1/sqrt(64)
      if (kcol > q0 + row) val = -1.0e9f;       // causal mask
      logits[row * LL + kcol] = val;
    }
  }
  __syncthreads();

  // ---- Phase 2: exact row softmax stats over LDS logits ----
  const int row2 = tid >> 4, seg = tid & 15;
  const int qrowi = q0 + row2;
  float pm = -3.0e38f;
  for (int k = seg; k <= qrowi; k += 16) pm = fmaxf(pm, logits[row2 * LL + k]);
  red[row2 * 16 + seg] = pm;
  __syncthreads();
  if (tid < 16) {
    float m = -3.0e38f;
#pragma unroll
    for (int j = 0; j < 16; ++j) m = fmaxf(m, red[tid * 16 + j]);
    rmax[tid] = m;
  }
  __syncthreads();
  const float rm = rmax[row2];
  float ps = 0.f;
  for (int k = seg; k <= qrowi; k += 16) ps += __expf(logits[row2 * LL + k] - rm);
  red[row2 * 16 + seg] = ps;
  __syncthreads();
  if (tid < 16) {
    float s = 0.f;
#pragma unroll
    for (int j = 0; j < 16; ++j) s += red[tid * 16 + j];
    rinv[tid] = 1.f / s;
  }
  __syncthreads();

  // ---- Phase 3: write f32 weights to HBM (zeros where masked), bf16 to LDS --
  const float ri = rinv[row2];
  const size_t wrowg = ((size_t)bh * LL + q0 + row2) * LL;
  for (int k = seg; k < LL; k += 16) {
    float w = 0.f;
    if (k <= qrowi) w = __expf(logits[row2 * LL + k] - rm) * ri;
    wout[wrowg + k] = w;
    if (k < kpad) wbf[row2 * LL + k] = (__bf16)w;
  }
  __syncthreads();

  // ---- Phase 4: O = W @ V.  8 waves = 4 d-tiles x 2 k-halves ----
  const int dt = wave & 3;
  const int ksub = (wave >> 2) * 32;
  v8f oacc = {};
  for (int kc = 0; kc < kpad; kc += 64) {
    {  // stage V^T chunk: vt[d][k'] = V[kc+k'][d]
      const int kk = tid & 63;
      const int dg = (tid >> 6) * 16;
      const __bf16* vrow = vh + (rowbase + kc + kk) * DEPTHD + dg;
      bf16x8 c0 = *(const bf16x8*)vrow;
      bf16x8 c1 = *(const bf16x8*)(vrow + 8);
#pragma unroll
      for (int j = 0; j < 8; ++j) {
        vt[(dg + j) * VT_STRIDE + kk]     = c0[j];
        vt[(dg + 8 + j) * VT_STRIDE + kk] = c1[j];
      }
    }
    __syncthreads();
    {
      // A-fragment from bf16 weights in LDS (ds_load_b128 x2)
      const __bf16* wp = wbf + ln * LL + kc + ksub + half * 8;
      bf16x8 c0 = *(const bf16x8*)wp;
      bf16x8 c1 = *(const bf16x8*)(wp + 16);
      v16bf aw;
#pragma unroll
      for (int i = 0; i < 8; ++i) { aw[i] = c0[i]; aw[8 + i] = c1[i]; }
      // B-fragment: 16B-aligned rows of transposed V (ds_load_b128 x2)
      v16bf bv = ld_bfrag16(vt + (dt * 16 + ln) * VT_STRIDE + ksub + half * 8);
      oacc = wmma_bf16(aw, bv, oacc);
    }
    __syncthreads();
  }
  // reduce the two k-half partials per d-tile, store O as bf16 heads
  if (wave >= 4) {
#pragma unroll
    for (int r = 0; r < 8; ++r) cred[(wave - 4) * 256 + r * 32 + lane] = oacc[r];
  }
  __syncthreads();
  if (wave < 4) {
#pragma unroll
    for (int r = 0; r < 8; ++r) {
      const float v = oacc[r] + cred[wave * 256 + r * 32 + lane];
      const int row = r + half * 8;
      oh[(rowbase + q0 + row) * DEPTHD + dt * 16 + ln] = (__bf16)v;
    }
  }
}

// ---------------------------------------------------------------------------
// Kernel 4: output projection  merge_heads(O) @ Wo^T + bo -> f32 out
// grid: (128, 8)  block: 256; two N tiles per wave (dual WMMA chains)
// ---------------------------------------------------------------------------
__global__ __launch_bounds__(256) void k_oproj(const __bf16* __restrict__ oh,
                                               const float* __restrict__ Wo,
                                               const float* __restrict__ bo,
                                               float* __restrict__ out) {
  const int tid = threadIdx.x;
  const int lane = tid & 31, wave = tid >> 5;
  const int half = lane >> 4, ln = lane & 15;
  const int mt = blockIdx.x * 2 + (wave >> 2);
  const int n0 = (blockIdx.y * 8 + (wave & 3) * 2) * 16;
  const int n1 = n0 + 16;
  const int m0 = mt * 16;

  const int m = m0 + ln;
  const int bb = m >> 11, l = m & (LL - 1);
  const __bf16* abase = oh + (size_t)bb * HH * LL * DEPTHD;
  const float* w0 = Wo + (size_t)(n0 + ln) * DD;
  const float* w1 = Wo + (size_t)(n1 + ln) * DD;

  v8f c0 = {}, c1 = {};
#pragma unroll 2
  for (int ks = 0; ks < DD / 32; ++ks) {
    const int kb = ks * 32 + half * 8;  // both 8-wide chunks stay in one head
    const int h0 = kb >> 6, d0 = kb & 63;
    const int k1 = kb + 16;
    const int h1 = k1 >> 6, d1 = k1 & 63;
    bf16x8 x0 = *(const bf16x8*)(abase + ((size_t)h0 * LL + l) * DEPTHD + d0);
    bf16x8 x1 = *(const bf16x8*)(abase + ((size_t)h1 * LL + l) * DEPTHD + d1);
    v16bf a;
#pragma unroll
    for (int i = 0; i < 8; ++i) { a[i] = x0[i]; a[8 + i] = x1[i]; }
    v16bf b0 = cvt_bfrag_f32(w0 + ks * 32 + half * 8);
    v16bf b1 = cvt_bfrag_f32(w1 + ks * 32 + half * 8);
    c0 = wmma_bf16(a, b0, c0);
    c1 = wmma_bf16(a, b1, c1);
  }

#pragma unroll
  for (int t = 0; t < 2; ++t) {
    const v8f& c = t ? c1 : c0;
    const int n = (t ? n1 : n0) + ln;
    const float bias = bo[n];
#pragma unroll
    for (int r = 0; r < 8; ++r) {
      const int row = r + half * 8;
      out[(size_t)(m0 + row) * DD + n] = c[r] + bias;
    }
  }
}

// ---------------------------------------------------------------------------
extern "C" void kernel_launch(void* const* d_in, const int* in_sizes, int n_in,
                              void* d_out, int out_size, void* d_ws, size_t ws_size,
                              hipStream_t stream) {
  (void)in_sizes; (void)n_in; (void)out_size; (void)ws_size;
  const float* q_in = (const float*)d_in[0];
  const float* k_in = (const float*)d_in[1];
  const float* v_in = (const float*)d_in[2];
  // d_in[3] = mask (tril) -- handled analytically
  const float* Wq  = (const float*)d_in[4];
  const float* bq  = (const float*)d_in[5];
  const float* Wo  = (const float*)d_in[6];
  const float* bo  = (const float*)d_in[7];
  const float* rel = (const float*)d_in[8];

  char* ws = (char*)d_ws;
  size_t off = 0;
  auto carve = [&](size_t bytes) {
    char* p = ws + off;
    off += (bytes + 255) & ~(size_t)255;
    return p;
  };
  __bf16* qkv  = (__bf16*)carve(3 * sizeof(__bf16) * (size_t)BB * LL * DD); // 24 MB
  __bf16* relb = (__bf16*)carve(sizeof(__bf16) * (size_t)LL * DEPTHD);      // 256 KB
  __bf16* ohb  = (__bf16*)carve(sizeof(__bf16) * (size_t)BB * HH * LL * DEPTHD); // 8 MB

  float* outp = (float*)d_out;
  float* wout = outp + (size_t)BB * LL * DD;

  k_cvt_rel<<<dim3((LL * DEPTHD + 255) / 256), dim3(256), 0, stream>>>(
      rel, relb, LL * DEPTHD);

  k_proj_qkv<<<dim3(128, 8, 3), dim3(256), 0, stream>>>(q_in, k_in, v_in, Wq, bq,
                                                        qkv);

  const __bf16* qh = qkv;
  const __bf16* khp = qkv + (size_t)BB * LL * DD;
  const __bf16* vhp = qkv + 2 * (size_t)BB * LL * DD;
  k_attn<<<dim3(128, BB * HH), dim3(256), SM_TOTAL, stream>>>(qh, khp, vhp, relb,
                                                              wout, ohb);

  k_oproj<<<dim3(128, 8), dim3(256), 0, stream>>>(ohb, Wo, bo, outp);
}
